// LSTMDecoder_2602750181927
// MI455X (gfx1250) — matrix operand order repaired and checked by reading, not compiled, not run on hardware
//
#include <hip/hip_runtime.h>

typedef __attribute__((ext_vector_type(16))) _Float16 v16h;
typedef __attribute__((ext_vector_type(8)))  _Float16 v8h;
typedef __attribute__((ext_vector_type(8)))  float    v8f;
typedef __attribute__((ext_vector_type(4)))  float    v4f;

#define HID   256
#define BATCH 2048
#define NG    1024            // 4*H gate width
#define MWG   16              // batch rows per workgroup
#define NTPW  4               // N-tiles per wave (64 tiles / 16 waves)
#define KT    8               // K tiles: 256 / 32
#define HSTR  264             // padded h row stride (f16): 528B = 4 dwords mod 64 banks
#define GSTR  1032            // padded gates row stride (f16)

// ---------------------------------------------------------------------------
// Pack W_ih and (W_ih + W_hh) into per-lane WMMA B-fragment order (f16), and
// fold biases. Packed element e: fragment f = e/512 (f = nt*8 + kt), lane
// l = (e%512)/16, element d2 = e%16.  B[k][n] = W[n][k];
//   n = nt*16 + (l&15),  k = kt*32 + (l>>4)*16 + d2.
// ---------------------------------------------------------------------------
__global__ void lstm_pack_kernel(const float* __restrict__ Wih,
                                 const float* __restrict__ Whh,
                                 const float* __restrict__ bih,
                                 const float* __restrict__ bhh,
                                 _Float16* __restrict__ pWih,
                                 _Float16* __restrict__ pWsum,
                                 float* __restrict__ bias)
{
    int e  = blockIdx.x * 256 + threadIdx.x;   // 0 .. 262143
    int f  = e >> 9;
    int r  = e & 511;
    int l  = r >> 4;
    int d2 = r & 15;
    int nt = f >> 3;
    int kt = f & 7;
    int n  = nt * 16 + (l & 15);
    int k  = kt * 32 + (l >> 4) * 16 + d2;
    float wi = Wih[n * HID + k];
    pWih[e]  = (_Float16)wi;
    pWsum[e] = (_Float16)(wi + Whh[n * HID + k]);
    if (e < NG) bias[e] = bih[e] + bhh[e];
}

__device__ __forceinline__ float sig_(float x) {
    return __builtin_amdgcn_rcpf(1.0f + __expf(-x));
}
__device__ __forceinline__ float tanh_(float x) {
    float e = __expf(-2.0f * x);
    return (1.0f - e) * __builtin_amdgcn_rcpf(1.0f + e);
}

// ---------------------------------------------------------------------------
// Persistent LSTM: each block owns 16 batch rows for all T steps.
// Wave w computes gate columns [w*64, w*64+64): tiles nt0+0/1 from 128
// persistent VGPRs, tiles nt0+2/3 from LDS (W_sum fully on-WGP resident).
// Elementwise phase: wave w owns batch row w; lane owns 8 consecutive cols.
// ---------------------------------------------------------------------------
__global__ __launch_bounds__(512, 1)
void lstm_main_kernel(const float* __restrict__ h_in,
                      const _Float16* __restrict__ pWih,
                      const _Float16* __restrict__ pWsum,
                      const float* __restrict__ bias,
                      const int* __restrict__ seqlen,
                      float* __restrict__ out)
{
    __shared__ _Float16 wcache[32 * KT * 512];   // 256 KB: 32 B-tiles of W_sum
    __shared__ _Float16 h_lds[MWG * HSTR];       // 8.25 KB (padded rows)
    __shared__ _Float16 gates_lds[MWG * GSTR];   // 32.25 KB (padded rows)

    const int tid  = threadIdx.x;
    const int lane = tid & 31;
    const int wid  = tid >> 5;
    const int lm   = lane & 15;     // row (A) / col (B,C) within tile
    const int lh   = lane >> 4;     // K-half (A) / row-half (C)
    const int nt0  = wid * NTPW;
    const int r0   = blockIdx.x * MWG;
    const int T    = seqlen[0];

    float* xhat = out;
    float* hf   = out + (size_t)T * BATCH * HID;
    float* cf   = hf + (size_t)BATCH * HID;

    // --- preload h (fp32 -> f16, padded rows) ---
    for (int i = tid; i < MWG * HID; i += 512)
        h_lds[(i >> 8) * HSTR + (i & 255)] =
            (_Float16)h_in[(size_t)(r0 + (i >> 8)) * HID + (i & 255)];

    // --- per-lane bias for the 4 owned gate columns (folded at scatter) ---
    float bias_q[NTPW];
    #pragma unroll
    for (int q = 0; q < NTPW; ++q)
        bias_q[q] = bias[(nt0 + q) * 16 + lm];

    // --- persistent register-resident B-tiles (nt0+0, nt0+1): 128 VGPRs ---
    v16h breg0[KT], breg1[KT];
    #pragma unroll
    for (int kt = 0; kt < KT; ++kt) {
        breg0[kt] = *(const v16h*)(pWsum +
            (((size_t)(nt0 + 0) * KT + kt) << 9) + (lane << 4));
        breg1[kt] = *(const v16h*)(pWsum +
            (((size_t)(nt0 + 1) * KT + kt) << 9) + (lane << 4));
    }

    // --- LDS-resident B-tiles (nt0+2, nt0+3) -> slots 2*wid, 2*wid+1 ---
    #pragma unroll
    for (int q = 0; q < 2; ++q) {
        const _Float16* src = pWsum + (((size_t)(nt0 + 2 + q) * KT) << 9);
        _Float16* dst = &wcache[(2 * wid + q) * (KT * 512)];
        for (int i = lane * 8; i < KT * 512; i += 32 * 8)
            *(v8h*)(dst + i) = *(const v8h*)(src + i);
    }

    float creg[8];
    #pragma unroll
    for (int i = 0; i < 8; ++i) creg[i] = 0.0f;

    __syncthreads();

    const _Float16* hrow = &h_lds[lm * HSTR];

#define A_LOAD(ktv)                                                           \
    v8h alo = *(const v8h*)(hrow + (ktv) * 32 + lh * 16);                     \
    v8h ahi = *(const v8h*)(hrow + (ktv) * 32 + lh * 16 + 8);                 \
    v16h a  = __builtin_shufflevector(alo, ahi,                               \
                  0,1,2,3,4,5,6,7,8,9,10,11,12,13,14,15);

#define WMMA_STEP(q, BSRC)                                                    \
    acc[q] = __builtin_amdgcn_wmma_f32_16x16x32_f16(                          \
        false, a, false, (BSRC), (short)0, acc[q], false, false);

    for (int t = 0; t < T; ++t) {
        v8f acc[NTPW] = {};

        if (t == 0) {                       // gates = x @ W_ih^T  (global/L2)
            #pragma unroll
            for (int kt = 0; kt < KT; ++kt) {
                A_LOAD(kt)
                #pragma unroll
                for (int q = 0; q < NTPW; ++q) {
                    v16h b = *(const v16h*)(pWih +
                        (((size_t)(nt0 + q) * KT + kt) << 9) + (lane << 4));
                    WMMA_STEP(q, b)
                }
            }
        } else {                            // W_sum fully VGPR/LDS resident
            #pragma unroll
            for (int kt = 0; kt < KT; ++kt) {
                A_LOAD(kt)
                WMMA_STEP(0, breg0[kt])
                WMMA_STEP(1, breg1[kt])
                v16h b2 = *(const v16h*)&wcache[
                    (2 * wid + 0) * (KT * 512) + (kt << 9) + (lane << 4)];
                WMMA_STEP(2, b2)
                v16h b3 = *(const v16h*)&wcache[
                    (2 * wid + 1) * (KT * 512) + (kt << 9) + (lane << 4)];
                WMMA_STEP(3, b3)
            }
        }

        // scatter biased C tiles into gates LDS (f16): m = v+lh*8, col = nt*16+lm
        #pragma unroll
        for (int q = 0; q < NTPW; ++q) {
            int col = (nt0 + q) * 16 + lm;
            #pragma unroll
            for (int v = 0; v < 8; ++v)
                gates_lds[(v + lh * 8) * GSTR + col] =
                    (_Float16)(acc[q][v] + bias_q[q]);
        }

        __syncthreads();

        // elementwise cell update: wave owns row wid, lane owns 8 consecutive
        // columns.  4x ds_load_b128 gathers all four (pre-biased) gates.
        const size_t tout = (t == 0) ? 0 : (size_t)(T - t);
        const bool last = (t == T - 1);

        const _Float16* grow = &gates_lds[wid * GSTR + lane * 8];
        v8h g0 = *(const v8h*)(grow + 0);
        v8h g1 = *(const v8h*)(grow + 256);
        v8h g2 = *(const v8h*)(grow + 512);
        v8h g3 = *(const v8h*)(grow + 768);

        v8h hnew;
        float hvf[8];
        #pragma unroll
        for (int i = 0; i < 8; ++i) {
            float ig = sig_((float)g0[i]);
            float fg = sig_((float)g1[i]);
            float gg = tanh_((float)g2[i]);
            float og = sig_((float)g3[i]);
            float c  = fg * creg[i] + ig * gg;
            creg[i]  = c;
            float hv = og * tanh_(c);
            hnew[i]  = (_Float16)hv;
            hvf[i]   = hv;
        }
        *(v8h*)&h_lds[wid * HSTR + lane * 8] = hnew;

        {
            v4f lo, hi;
            #pragma unroll
            for (int i = 0; i < 4; ++i) { lo[i] = hvf[i]; hi[i] = hvf[i + 4]; }
            float* orow = xhat + (tout * BATCH + (size_t)(r0 + wid)) * HID + lane * 8;
            __builtin_nontemporal_store(lo, (v4f*)orow);
            __builtin_nontemporal_store(hi, (v4f*)(orow + 4));
            if (last) {
                float* hrp = hf + (size_t)(r0 + wid) * HID + lane * 8;
                float* crp = cf + (size_t)(r0 + wid) * HID + lane * 8;
                v4f cl, ch;
                #pragma unroll
                for (int i = 0; i < 4; ++i) { cl[i] = creg[i]; ch[i] = creg[i + 4]; }
                __builtin_nontemporal_store(lo, (v4f*)hrp);
                __builtin_nontemporal_store(hi, (v4f*)(hrp + 4));
                __builtin_nontemporal_store(cl, (v4f*)crp);
                __builtin_nontemporal_store(ch, (v4f*)(crp + 4));
            }
        }
        __syncthreads();
    }
#undef A_LOAD
#undef WMMA_STEP
}

extern "C" void kernel_launch(void* const* d_in, const int* in_sizes, int n_in,
                              void* d_out, int out_size, void* d_ws, size_t ws_size,
                              hipStream_t stream)
{
    const float* h_in = (const float*)d_in[0];
    const float* Wih  = (const float*)d_in[1];
    const float* Whh  = (const float*)d_in[2];
    const float* bih  = (const float*)d_in[3];
    const float* bhh  = (const float*)d_in[4];
    const int*   seql = (const int*)d_in[5];

    _Float16* pWih  = (_Float16*)d_ws;                 // 512 KB
    _Float16* pWsum = pWih + NG * HID;                 // 512 KB
    float*    bias  = (float*)(pWsum + NG * HID);      // 4 KB

    lstm_pack_kernel<<<NG * HID / 256, 256, 0, stream>>>(
        Wih, Whh, bih, bhh, pWih, pWsum, bias);
    lstm_main_kernel<<<BATCH / MWG, 512, 0, stream>>>(
        h_in, pWih, pWsum, bias, seql, (float*)d_out);
}
